// GraphTransformerEmbedder_18889266167949
// MI455X (gfx1250) — matrix operand order repaired and checked by reading, not compiled, not run on hardware
//
#include <hip/hip_runtime.h>

#define NNODES  32768
#define NEDGES  524288
#define HIDD    128
#define NHEADS  4
#define DHEAD   32
#define NLAYERS 3
#define NPER    2048
#define BGRAPH  16

typedef __attribute__((ext_vector_type(16))) _Float16 v16h;
typedef __attribute__((ext_vector_type(8)))  _Float16 v8h;
typedef __attribute__((ext_vector_type(8)))  float    v8f;
typedef __attribute__((ext_vector_type(4)))  unsigned v4u;
typedef __attribute__((ext_vector_type(8)))  int      v8i;
typedef __attribute__((ext_vector_type(4)))  int      v4i;

#if defined(__gfx1250__) && __has_builtin(__builtin_amdgcn_tensor_load_to_lds)
#define USE_TDM 1
#else
#define USE_TDM 0
#endif

// ---- order-preserving float <-> uint mapping for atomic max ----
__device__ __forceinline__ unsigned enc_f32(float f) {
    unsigned u = __float_as_uint(f);
    return (u & 0x80000000u) ? ~u : (u | 0x80000000u);
}
__device__ __forceinline__ float dec_f32(unsigned k) {
    return __uint_as_float((k & 0x80000000u) ? (k & 0x7fffffffu) : ~k);
}

#if USE_TDM
// Issue one TDM 2-D tile load (rows x cols f16, row stride == cols) into LDS.
// Must be executed by ONE wave; caller barriers afterwards.
// clang-23 lane: 6-arg builtin (v4u g0, v8i g1, v4i g2, v4i g3, v8i, i32 cpol).
__device__ __forceinline__ void tdm_load_2d_f16(const _Float16* gsrc, _Float16* lds_dst,
                                                int rows, int cols) {
    unsigned long long ga = (unsigned long long)(uintptr_t)gsrc;
    unsigned lds = (unsigned)(uintptr_t)lds_dst;     // flat LDS addr: offset in low 32 bits
    v4u g0;
    g0[0] = 1u;                                       // count=1, user descriptor
    g0[1] = lds;                                      // lds_addr
    g0[2] = (unsigned)(ga & 0xffffffffu);             // global_addr[31:0]
    g0[3] = (unsigned)((ga >> 32) & 0x01ffffffu) | (2u << 30);  // addr[56:32] | type=2
    v8i g1;
    g1[0] = (int)(1u << 16);                          // data_size=1 (2 bytes), no flags
    g1[1] = (int)((unsigned)(cols & 0xffff) << 16);   // tensor_dim0[15:0]
    g1[2] = (int)(((unsigned)cols >> 16) | ((unsigned)(rows & 0xffff) << 16)); // d0 hi | d1 lo
    g1[3] = (int)(((unsigned)rows >> 16) | ((unsigned)(cols & 0xffff) << 16)); // d1 hi | tile_dim0
    g1[4] = (int)(unsigned)(rows & 0xffff);           // tile_dim1 (tile_dim2=0)
    g1[5] = cols;                                     // tensor_dim0_stride[31:0]
    g1[6] = 0;
    g1[7] = 0;
    v4i gz4 = {0, 0, 0, 0};
    v8i gz8 = {0, 0, 0, 0, 0, 0, 0, 0};
    __builtin_amdgcn_tensor_load_to_lds(g0, g1, gz4, gz4, gz8, 0);
    __builtin_amdgcn_s_wait_tensorcnt(0);
}
#endif

// Stage `rows`x128 f16 weight block into LDS (TDM if available, else cooperative copy).
__device__ __forceinline__ void stage_weights(const _Float16* __restrict__ Wt,
                                              _Float16* lw, int rows) {
#if USE_TDM
    if (threadIdx.x < 32) tdm_load_2d_f16(Wt, lw, rows, HIDD);
    __syncthreads();
#else
    const float4* src = (const float4*)Wt;            // rows*128*2/16 float4s
    float4* dst = (float4*)lw;
    const int n16 = rows * HIDD / 8;
    for (int i = threadIdx.x; i < n16; i += 256) dst[i] = src[i];
    __syncthreads();
#endif
}

// ---------------- utility kernels ----------------
__global__ void fill_u32(unsigned* p, unsigned v, int n) {
    int i = blockIdx.x * blockDim.x + threadIdx.x;
    if (i < n) p[i] = v;
}

__global__ void embed_kernel(const float* __restrict__ x,
                             const float* __restrict__ in_w,
                             const float* __restrict__ in_b,
                             float* __restrict__ h) {
    int i = blockIdx.x * blockDim.x + threadIdx.x;    // N*HID threads
    int n = i >> 7, c = i & 127;
    float a = x[n * 3 + 0] * in_w[0 * HIDD + c]
            + x[n * 3 + 1] * in_w[1 * HIDD + c]
            + x[n * 3 + 2] * in_w[2 * HIDD + c] + in_b[c];
    h[i] = a;
}

// transpose + convert one 128x128 weight: wt[n][k] = (f16) w[k][n]
__global__ void prep_weight(const float* __restrict__ w, _Float16* __restrict__ wt) {
    int i = blockIdx.x * blockDim.x + threadIdx.x;    // 16384 threads
    int n = i >> 7, k = i & 127;
    wt[n * HIDD + k] = (_Float16)w[k * HIDD + n];
}

__global__ void cvt_f16(const float* __restrict__ s, _Float16* __restrict__ d, int n) {
    int i = blockIdx.x * blockDim.x + threadIdx.x;
    if (i < n) d[i] = (_Float16)s[i];
}

// ---- shared WMMA helpers ----
// A fragment: lanes 0-15 row=lane, K {kb..kb+15}; lanes 16-31 K +16.
__device__ __forceinline__ v16h load_a_frag(const _Float16* ap, int kb) {
    v16h a;
    *((v8h*)&a)     = *(const v8h*)(ap + kb);
    *((v8h*)&a + 1) = *(const v8h*)(ap + kb + 8);
    return a;
}

// ---------------- fused QKV WMMA GEMM ----------------
// Wt3 = [Wq^T | Wk^T | Wv^T] contiguous, 3*128*128 f16 staged in 96 KB LDS.
// Block: 8 waves, tile 64 rows x 128 cols; wave: 16 rows x 64 cols x 3 outputs.
__global__ __launch_bounds__(256)
void gemm_qkv(const _Float16* __restrict__ A, const _Float16* __restrict__ Wt3,
              const float* __restrict__ bq, const float* __restrict__ bk,
              const float* __restrict__ bv,
              float* __restrict__ oq, float* __restrict__ ok, float* __restrict__ ov) {
    __shared__ _Float16 lw[3 * HIDD * HIDD];          // 96 KB of 320 KB WGP LDS
    stage_weights(Wt3, lw, 3 * HIDD);

    const int wave   = threadIdx.x >> 5;
    const int lane   = threadIdx.x & 31;
    const int laneLo = lane & 15;
    const int laneHi = lane >> 4;
    const int row0    = blockIdx.x * 64 + (wave >> 1) * 16;
    const int colBase = (wave & 1) * 64;

    const _Float16* ap = A + (size_t)(row0 + laneLo) * HIDD + laneHi * 16;

    v8f acc[3][4] = {};
#pragma unroll
    for (int kc = 0; kc < 4; ++kc) {
        const int kb = kc * 32;
        const v16h a = load_a_frag(ap, kb);
#pragma unroll
        for (int m = 0; m < 3; ++m) {
#pragma unroll
            for (int nt = 0; nt < 4; ++nt) {
                const int col = colBase + nt * 16 + laneLo;
                const v16h b = *(const v16h*)(lw + m * HIDD * HIDD + col * HIDD + laneHi * 16 + kb);
                acc[m][nt] = __builtin_amdgcn_wmma_f32_16x16x32_f16(
                    false, a, false, b, (short)0, acc[m][nt], false, false);
            }
        }
    }

    const float* bs[3] = {bq, bk, bv};
    float* os[3] = {oq, ok, ov};
#pragma unroll
    for (int m = 0; m < 3; ++m) {
#pragma unroll
        for (int nt = 0; nt < 4; ++nt) {
            const int col = colBase + nt * 16 + laneLo;
            const float bias = bs[m][col];
#pragma unroll
            for (int r = 0; r < 8; ++r) {
                const int row = row0 + r + laneHi * 8;
                os[m][(size_t)row * HIDD + col] = acc[m][nt][r] + bias;
            }
        }
    }
}

// ---------------- single WMMA GEMM: out[M,128] = A @ W + bias (+fusions) ----------------
// MODE 1: out = add + acc+bias ; MODE 2: out = add + relu(acc+bias)
template <int MODE>
__global__ __launch_bounds__(256)
void gemm128(const _Float16* __restrict__ A, const _Float16* __restrict__ Wt,
             const float* __restrict__ bias, const float* __restrict__ add,
             float* __restrict__ out) {
    __shared__ _Float16 lw[HIDD * HIDD];              // 32 KB
    stage_weights(Wt, lw, HIDD);

    const int wave   = threadIdx.x >> 5;
    const int lane   = threadIdx.x & 31;
    const int laneLo = lane & 15;
    const int laneHi = lane >> 4;
    const int row0    = blockIdx.x * 64 + (wave >> 1) * 16;
    const int colBase = (wave & 1) * 64;

    const _Float16* ap = A + (size_t)(row0 + laneLo) * HIDD + laneHi * 16;

    v8f acc[4] = {};
#pragma unroll
    for (int kc = 0; kc < 4; ++kc) {
        const int kb = kc * 32;
        const v16h a = load_a_frag(ap, kb);
#pragma unroll
        for (int nt = 0; nt < 4; ++nt) {
            const int col = colBase + nt * 16 + laneLo;
            const v16h b = *(const v16h*)(lw + col * HIDD + laneHi * 16 + kb);
            acc[nt] = __builtin_amdgcn_wmma_f32_16x16x32_f16(
                false, a, false, b, (short)0, acc[nt], false, false);
        }
    }

#pragma unroll
    for (int nt = 0; nt < 4; ++nt) {
        const int col = colBase + nt * 16 + laneLo;
        const float bs = bias[col];
#pragma unroll
        for (int r = 0; r < 8; ++r) {
            const int row = row0 + r + laneHi * 8;
            const size_t idx = (size_t)row * HIDD + col;
            float v = acc[nt][r] + bs;
            if (MODE == 2) v = fmaxf(v, 0.0f);
            v += add[idx];
            out[idx] = v;
        }
    }
}

// ---------------- edge attention kernels ----------------
__global__ void edge_logits(const float* __restrict__ q, const float* __restrict__ k,
                            const int* __restrict__ esrc, const int* __restrict__ edst,
                            float* __restrict__ logits, unsigned* __restrict__ mkey) {
    int t = blockIdx.x * blockDim.x + threadIdx.x;    // E*HEADS
    if (t >= NEDGES * NHEADS) return;
    int e = t >> 2, hd = t & 3;
    int s = esrc[e], d = edst[e];
    const float4* qp = (const float4*)(q + (size_t)d * HIDD + hd * DHEAD);
    const float4* kp = (const float4*)(k + (size_t)s * HIDD + hd * DHEAD);
    float acc = 0.f;
#pragma unroll
    for (int i = 0; i < 8; ++i) {
        float4 a = qp[i], b = kp[i];
        acc += a.x * b.x + a.y * b.y + a.z * b.z + a.w * b.w;
    }
    acc *= 0.17677669529663689f;                      // 1/sqrt(32)
    logits[t] = acc;
    atomicMax(&mkey[d * NHEADS + hd], enc_f32(acc));
}

__global__ void edge_exp(float* __restrict__ ebuf, const int* __restrict__ edst,
                         const unsigned* __restrict__ mkey, float* __restrict__ z) {
    int t = blockIdx.x * blockDim.x + threadIdx.x;
    if (t >= NEDGES * NHEADS) return;
    int e = t >> 2, hd = t & 3;
    int d = edst[e];
    float m = dec_f32(mkey[d * NHEADS + hd]);
    float ex = __expf(ebuf[t] - m);
    ebuf[t] = ex;                                     // in-place logits -> exp
    atomicAdd(&z[d * NHEADS + hd], ex);
}

__global__ void edge_agg(const float* __restrict__ v, const int* __restrict__ esrc,
                         const int* __restrict__ edst, const float* __restrict__ ebuf,
                         const float* __restrict__ z, float* __restrict__ agg) {
    int t = blockIdx.x * blockDim.x + threadIdx.x;    // E*HID = 64M < 2^31
    if (t >= NEDGES * HIDD) return;
    int e = t >> 7, c = t & 127, hd = c >> 5;
    int s = esrc[e], d = edst[e];
    float alpha = ebuf[(size_t)e * NHEADS + hd] / z[d * NHEADS + hd];
    atomicAdd(&agg[(size_t)d * HIDD + c], alpha * v[(size_t)s * HIDD + c]);
}

// ---------------- layernorm( a + res ) -> relu ; one wave32 per node ----------------
__global__ void ln_relu(const float* __restrict__ a, const float* __restrict__ res,
                        const float* __restrict__ g, const float* __restrict__ bta,
                        float* __restrict__ out) {
    int n = blockIdx.x, lane = threadIdx.x;           // 32 lanes x 4 channels
    size_t base = (size_t)n * HIDD + lane * 4;
    float4 av = *(const float4*)(a + base);
    float4 rv = *(const float4*)(res + base);
    float v0 = av.x + rv.x, v1 = av.y + rv.y, v2 = av.z + rv.z, v3 = av.w + rv.w;
    float s = v0 + v1 + v2 + v3;
    for (int off = 16; off; off >>= 1) s += __shfl_xor(s, off, 32);
    float mu = s * (1.0f / HIDD);
    float d0 = v0 - mu, d1 = v1 - mu, d2 = v2 - mu, d3 = v3 - mu;
    float vs = d0 * d0 + d1 * d1 + d2 * d2 + d3 * d3;
    for (int off = 16; off; off >>= 1) vs += __shfl_xor(vs, off, 32);
    float inv = rsqrtf(vs * (1.0f / HIDD) + 1e-5f);
    int c = lane * 4;
    float4 o;
    o.x = fmaxf(d0 * inv * g[c + 0] + bta[c + 0], 0.f);
    o.y = fmaxf(d1 * inv * g[c + 1] + bta[c + 1], 0.f);
    o.z = fmaxf(d2 * inv * g[c + 2] + bta[c + 2], 0.f);
    o.w = fmaxf(d3 * inv * g[c + 3] + bta[c + 3], 0.f);
    *(float4*)(out + base) = o;
}

// ---------------- mean pool per graph (nodes are contiguous) ----------------
__global__ void pool_mean(const float* __restrict__ h, float* __restrict__ out) {
    int b = blockIdx.x, c = threadIdx.x;              // 16 x 128
    const float* p = h + (size_t)b * NPER * HIDD + c;
    float s = 0.f;
    for (int n = 0; n < NPER; ++n) s += p[(size_t)n * HIDD];
    out[b * HIDD + c] = s * (1.0f / NPER);
}

extern "C" void kernel_launch(void* const* d_in, const int* in_sizes, int n_in,
                              void* d_out, int out_size, void* d_ws, size_t ws_size,
                              hipStream_t stream) {
    const float* x    = (const float*)d_in[0];
    const int*   esrc = (const int*)d_in[1];
    const int*   edst = (const int*)d_in[2];
    /* d_in[3] seg: unused (graphs are contiguous blocks of NPER nodes) */
    const float* in_w = (const float*)d_in[4];
    const float* in_b = (const float*)d_in[5];
    const float* qw = (const float*)d_in[6],  *qb = (const float*)d_in[7];
    const float* kw = (const float*)d_in[8],  *kb = (const float*)d_in[9];
    const float* vw = (const float*)d_in[10], *vb = (const float*)d_in[11];
    const float* sw = (const float*)d_in[12], *sb = (const float*)d_in[13];
    const float* fw = (const float*)d_in[14], *fb = (const float*)d_in[15];
    const float* lng = (const float*)d_in[16], *lnb = (const float*)d_in[17];

    char* ws = (char*)d_ws;
    size_t off = 0;
    auto alloc = [&](size_t bytes) -> void* {
        void* p = ws + off;
        off += (bytes + 255) & ~(size_t)255;
        return p;
    };
    const size_t HN = (size_t)NNODES * HIDD;
    float*    hA   = (float*)alloc(HN * 4);           // ping-pong node features
    float*    hB   = (float*)alloc(HN * 4);
    float*    q    = (float*)alloc(HN * 4);
    float*    k    = (float*)alloc(HN * 4);
    float*    v    = (float*)alloc(HN * 4);
    float*    agg  = (float*)alloc(HN * 4);
    float*    ebuf = (float*)alloc((size_t)NEDGES * NHEADS * 4);
    float*    z    = (float*)alloc((size_t)NNODES * NHEADS * 4);
    unsigned* mkey = (unsigned*)alloc((size_t)NNODES * NHEADS * 4);
    _Float16* h16  = (_Float16*)alloc(HN * 2);
    _Float16* w16  = (_Float16*)alloc((size_t)15 * HIDD * HIDD * 2);
    float* t1 = q;  // q free once attention aggregation is done
    float* t2 = k;  // k likewise

    const int HH = HIDD * HIDD;

    // transpose+convert all layer weights to f16 [n][k]; q,k,v contiguous per layer
    for (int l = 0; l < NLAYERS; ++l) {
        prep_weight<<<HH / 256, 256, 0, stream>>>(qw + l * HH, w16 + (size_t)(l * 5 + 0) * HH);
        prep_weight<<<HH / 256, 256, 0, stream>>>(kw + l * HH, w16 + (size_t)(l * 5 + 1) * HH);
        prep_weight<<<HH / 256, 256, 0, stream>>>(vw + l * HH, w16 + (size_t)(l * 5 + 2) * HH);
        prep_weight<<<HH / 256, 256, 0, stream>>>(sw + l * HH, w16 + (size_t)(l * 5 + 3) * HH);
        prep_weight<<<HH / 256, 256, 0, stream>>>(fw + l * HH, w16 + (size_t)(l * 5 + 4) * HH);
    }

    float* hcur = hA;
    float* hnxt = hB;
    embed_kernel<<<(int)(HN / 256), 256, 0, stream>>>(x, in_w, in_b, hcur);

    const int gGemm = NNODES / 64;                    // 512 blocks x 8 waves
    const int gEH   = (NEDGES * NHEADS) / 256;        // 8192
    const int gEC   = (NEDGES * HIDD) / 256;          // 262144

    for (int l = 0; l < NLAYERS; ++l) {
        cvt_f16<<<(int)(HN / 256), 256, 0, stream>>>(hcur, h16, (int)HN);

        // fused q,k,v projection (one pass over A, 3 weight tiles staged via TDM)
        gemm_qkv<<<gGemm, 256, 0, stream>>>(h16, w16 + (size_t)(l * 5) * HH,
                                            qb + l * HIDD, kb + l * HIDD, vb + l * HIDD,
                                            q, k, v);

        fill_u32<<<(NNODES * NHEADS) / 256, 256, 0, stream>>>(mkey, 0x007fffffu /* enc(-inf) */, NNODES * NHEADS);
        fill_u32<<<(NNODES * NHEADS) / 256, 256, 0, stream>>>((unsigned*)z, 0u, NNODES * NHEADS);
        fill_u32<<<(int)(HN / 256), 256, 0, stream>>>((unsigned*)agg, 0u, (int)HN);

        edge_logits<<<gEH, 256, 0, stream>>>(q, k, esrc, edst, ebuf, mkey);
        edge_exp<<<gEH, 256, 0, stream>>>(ebuf, edst, mkey, z);
        edge_agg<<<gEC, 256, 0, stream>>>(v, esrc, edst, ebuf, z, agg);

        // t1 = agg + h @ sw + sb
        gemm128<1><<<gGemm, 256, 0, stream>>>(h16, w16 + (size_t)(l * 5 + 3) * HH, sb + l * HIDD, agg, t1);
        // t2 = t1 + relu(t1 @ fw + fb)
        cvt_f16<<<(int)(HN / 256), 256, 0, stream>>>(t1, h16, (int)HN);
        gemm128<2><<<gGemm, 256, 0, stream>>>(h16, w16 + (size_t)(l * 5 + 4) * HH, fb + l * HIDD, t1, t2);
        // h_next = relu(LN(t2 + h_residual))
        ln_relu<<<NNODES, 32, 0, stream>>>(t2, hcur, lng + l * HIDD, lnb + l * HIDD, hnxt);

        float* tmp = hcur; hcur = hnxt; hnxt = tmp;
    }

    pool_mean<<<BGRAPH, HIDD, 0, stream>>>(hcur, (float*)d_out);
}
